// MoE_62027917689541
// MI455X (gfx1250) — matrix operand order repaired and checked by reading, not compiled, not run on hardware
//
#include <hip/hip_runtime.h>
#include <math.h>

// ---------------------------------------------------------------------------
// Model dims (match reference)
// ---------------------------------------------------------------------------
#define BTOK 8192
#define DIN  512
#define DH   1024
#define DOUT 512
#define NEXP 8
#define TAU1 0.05f
#define TAU2 0.01f
#define EPS_TK 0.5f
#define KSEL 2.0f
#define LOSS_COEF 0.01f

typedef __attribute__((ext_vector_type(16))) __bf16 v16bf;
typedef __attribute__((ext_vector_type(4)))  __bf16 v4bf;
typedef __attribute__((ext_vector_type(8)))  float  v8f;
typedef int v4i __attribute__((vector_size(16)));

union Frag { v16bf v; uint4 u[2]; };

// ---------------------------------------------------------------------------
// CDNA5 async global->LDS copy (16B per lane), with safe fallback.
// ---------------------------------------------------------------------------
#if defined(__gfx1250__) && __has_builtin(__builtin_amdgcn_global_load_async_to_lds_b128)
#define HAS_ASYNC_LDS 1
#endif

__device__ __forceinline__ void copy16_to_lds(const __bf16* gsrc, __bf16* ldst)
{
#ifdef HAS_ASYNC_LDS
    __builtin_amdgcn_global_load_async_to_lds_b128(
        (__attribute__((address_space(1))) v4i*)gsrc,
        (__attribute__((address_space(3))) v4i*)ldst, 0, 0);
#else
    *(uint4*)ldst = *(const uint4*)gsrc;
#endif
}

__device__ __forceinline__ void wait_async_copies()
{
#ifdef HAS_ASYNC_LDS
#if __has_builtin(__builtin_amdgcn_s_wait_asynccnt)
    __builtin_amdgcn_s_wait_asynccnt(0);
#else
    asm volatile("s_wait_asynccnt 0x0" ::: "memory");
#endif
#endif
}

// ---------------------------------------------------------------------------
// fp32 -> bf16 elementwise (x), 4 elems/thread
// ---------------------------------------------------------------------------
__global__ __launch_bounds__(256)
void cvt_bf16_kernel(const float* __restrict__ src, __bf16* __restrict__ dst, int n4)
{
    const int i = blockIdx.x * 256 + threadIdx.x;
    if (i < n4) {
        const float4 v = ((const float4*)src)[i];
        v4bf p = { (__bf16)v.x, (__bf16)v.y, (__bf16)v.z, (__bf16)v.w };
        ((v4bf*)dst)[i] = p;
    }
}

// ---------------------------------------------------------------------------
// W[K][N] fp32 -> Wt[N][K] bf16, 32x32 tiles (coalesced both sides)
// ---------------------------------------------------------------------------
__global__ __launch_bounds__(256)
void transpose_cvt_kernel(const float* __restrict__ W, __bf16* __restrict__ Wt,
                          int Kd, int N)
{
    __shared__ float tile[32][33];
    const int tx = threadIdx.x;      // 0..31
    const int ty = threadIdx.y;      // 0..7
    const int n0 = blockIdx.x * 32;
    const int k0 = blockIdx.y * 32;
#pragma unroll
    for (int i = 0; i < 4; ++i) {
        const int k = k0 + ty + i * 8;
        tile[ty + i * 8][tx] = W[(size_t)k * N + n0 + tx];
    }
    __syncthreads();
#pragma unroll
    for (int i = 0; i < 4; ++i) {
        const int n = n0 + ty + i * 8;
        Wt[(size_t)n * Kd + k0 + tx] = (__bf16)tile[tx][ty + i * 8];
    }
}

// ---------------------------------------------------------------------------
// Gating: one wave (32 lanes) per token.
// ---------------------------------------------------------------------------
__global__ __launch_bounds__(256)
void gating_kernel(const float* __restrict__ x,
                   const float* __restrict__ Wg1, const float* __restrict__ bg1,
                   const float* __restrict__ Wg2, const float* __restrict__ bg2,
                   float* __restrict__ gates)
{
    const int lane  = threadIdx.x & 31;
    const int wave  = threadIdx.x >> 5;
    const int token = blockIdx.x * 8 + wave;

    float acc[NEXP];
#pragma unroll
    for (int e = 0; e < NEXP; ++e) acc[e] = 0.f;

    const float* xr = x + (size_t)token * DIN;
    for (int k = lane; k < DIN; k += 32) {
        const float xv = xr[k];
        const float* wr = Wg1 + (size_t)k * NEXP;
#pragma unroll
        for (int e = 0; e < NEXP; ++e) acc[e] = fmaf(xv, wr[e], acc[e]);
    }
#pragma unroll
    for (int off = 16; off > 0; off >>= 1) {
#pragma unroll
        for (int e = 0; e < NEXP; ++e)
            acc[e] += __shfl_xor(acc[e], off, 32);
    }

    if (lane == 0) {
        float l1[NEXP], lg[NEXP], s[NEXP];
#pragma unroll
        for (int e = 0; e < NEXP; ++e) l1[e] = tanhf(acc[e] + bg1[e]);
#pragma unroll
        for (int e = 0; e < NEXP; ++e) {
            float v = bg2[e];
#pragma unroll
            for (int j = 0; j < NEXP; ++j) v = fmaf(l1[j], Wg2[j * NEXP + e], v);
            lg[e] = v;
        }
        float mx = lg[0];
#pragma unroll
        for (int e = 1; e < NEXP; ++e) mx = fmaxf(mx, lg[e]);
        float den = 0.f;
#pragma unroll
        for (int e = 0; e < NEXP; ++e) { s[e] = expf(lg[e] - mx); den += s[e]; }
        const float inv = 1.f / den;
#pragma unroll
        for (int e = 0; e < NEXP; ++e) s[e] *= inv;
        float* gr = gates + (size_t)token * NEXP;
#pragma unroll
        for (int i = 0; i < NEXP; ++i) {
            float sig = 0.f;
#pragma unroll
            for (int j = 0; j < NEXP; ++j)
                sig += 1.f / (1.f + expf((s[i] - s[j]) / TAU1));
            const float rt = 1.f + (sig - 0.5f);
            const float a  = 1.f / (1.f + expf((rt - (KSEL + EPS_TK)) / TAU2));
            gr[i] = a * s[i];
        }
    }
}

__global__ __launch_bounds__(256)
void importance_kernel(const float* __restrict__ gates, float* __restrict__ imp)
{
    __shared__ float red[256];
    const int e = blockIdx.x, t = threadIdx.x;
    float s = 0.f;
    for (int b = t; b < BTOK; b += 256) s += gates[(size_t)b * NEXP + e];
    red[t] = s;
    __syncthreads();
    for (int o = 128; o > 0; o >>= 1) {
        if (t < o) red[t] += red[t + o];
        __syncthreads();
    }
    if (t == 0) imp[e] = red[0];
}

__global__ void loss_kernel(const float* __restrict__ imp, float* __restrict__ out)
{
    float m = 0.f;
#pragma unroll
    for (int e = 0; e < NEXP; ++e) m += imp[e];
    m *= (1.f / NEXP);
    float v = 0.f;
#pragma unroll
    for (int e = 0; e < NEXP; ++e) { const float d = imp[e] - m; v += d * d; }
    v *= (1.f / (NEXP - 1));
    out[0] = LOSS_COEF * v / (m * m + 1e-10f);
}

__global__ __launch_bounds__(256)
void zero_kernel(float* __restrict__ y, int n)
{
    const int i = blockIdx.x * 256 + threadIdx.x;
    if (i < n) y[i] = 0.f;
}

// ---------------------------------------------------------------------------
// WMMA bf16 GEMM: C[M,N] = epilogue(A[M,K] @ Bt[N,K]^T + bias)
//   A: bf16 [M][Kd] (k-contiguous).  Bt: bf16 [N][Kd] (k-contiguous).
//   Block tile 128x128, BK=32, 256 threads = 8 waves (2x4), wave tile 64x32.
//   Double-buffered LDS: async DMA of tile i+1 overlaps WMMA on tile i.
//   EPI==0: Cout(bf16) = tanh(acc + bias)
//   EPI==1: y(fp32)   += gates[row][expert] * (acc + bias)
// ---------------------------------------------------------------------------
template <int EPI>
__global__ __launch_bounds__(256)
void gemm_wmma(const __bf16* __restrict__ A, const __bf16* __restrict__ Bt,
               const float* __restrict__ bias, int N, int Kd,
               void* __restrict__ Cout, const float* __restrict__ gates, int expert)
{
    __shared__ __align__(16) __bf16 Asm[2][128][40];  // [buf][m][k]
    __shared__ __align__(16) __bf16 Bsm[2][128][40];  // [buf][n][k]

    const int tid  = threadIdx.x;
    const int lane = tid & 31;
    const int wid  = tid >> 5;
    const int half = lane >> 4;
    const int r    = lane & 15;
    const int wmBase = (wid >> 2) * 64;  // 2 wave rows
    const int wnBase = (wid & 3) * 32;   // 4 wave cols

    const int n0 = blockIdx.x * 128;
    const int m0 = blockIdx.y * 128;

    // staging assignment: 16B per copy, 2 rows per thread per operand
    const int tcol = (tid & 3) * 8;   // 0..24
    const int trow = tid >> 2;        // 0..63

    v8f acc[4][2];
#pragma unroll
    for (int f = 0; f < 4; ++f)
#pragma unroll
        for (int g = 0; g < 2; ++g) acc[f][g] = (v8f)(0.f);

    auto issue_tile = [&](int buf, int kt) {
#pragma unroll
        for (int rr = 0; rr < 2; ++rr) {
            const int row = trow + rr * 64;
            copy16_to_lds(A  + (size_t)(m0 + row) * Kd + kt + tcol, &Asm[buf][row][tcol]);
            copy16_to_lds(Bt + (size_t)(n0 + row) * Kd + kt + tcol, &Bsm[buf][row][tcol]);
        }
    };

    const int nIter = Kd / 32;
    issue_tile(0, 0);

    for (int i = 0; i < nIter; ++i) {
        const int cur = i & 1;
        wait_async_copies();   // this wave's loads into buf `cur` are complete
        __syncthreads();       // all waves: buf `cur` ready; buf `cur^1` fully consumed
        if (i + 1 < nIter)
            issue_tile(cur ^ 1, (i + 1) * 32);  // DMA overlaps WMMAs below

        Frag af[4], bfr[2];
#pragma unroll
        for (int f = 0; f < 4; ++f) {
            const __bf16* p = &Asm[cur][wmBase + f * 16 + r][8 * half];
            af[f].u[0] = *(const uint4*)(p);
            af[f].u[1] = *(const uint4*)(p + 16);
        }
#pragma unroll
        for (int g = 0; g < 2; ++g) {
            const __bf16* p = &Bsm[cur][wnBase + g * 16 + r][8 * half];
            bfr[g].u[0] = *(const uint4*)(p);
            bfr[g].u[1] = *(const uint4*)(p + 16);
        }
#pragma unroll
        for (int f = 0; f < 4; ++f)
#pragma unroll
            for (int g = 0; g < 2; ++g)
                acc[f][g] = __builtin_amdgcn_wmma_f32_16x16x32_bf16(
                    false, af[f].v, false, bfr[g].v, (short)0, acc[f][g], false, false);
    }

#pragma unroll
    for (int f = 0; f < 4; ++f) {
#pragma unroll
        for (int g = 0; g < 2; ++g) {
            const int rowb = m0 + wmBase + f * 16 + 8 * half;
            const int gcol = n0 + wnBase + g * 16 + r;
            const float bv = bias[gcol];
            if (EPI == 0) {
                __bf16* C = (__bf16*)Cout;
#pragma unroll
                for (int v = 0; v < 8; ++v) {
                    const float val = tanhf(acc[f][g][v] + bv);
                    C[(size_t)(rowb + v) * N + gcol] = (__bf16)val;
                }
            } else {
                float* Y = (float*)Cout;
#pragma unroll
                for (int v = 0; v < 8; ++v) {
                    const int row = rowb + v;
                    const float val = acc[f][g][v] + bv;
                    Y[(size_t)row * N + gcol] += gates[(size_t)row * NEXP + expert] * val;
                }
            }
        }
    }
}

// ---------------------------------------------------------------------------
// Launch
// ---------------------------------------------------------------------------
extern "C" void kernel_launch(void* const* d_in, const int* in_sizes, int n_in,
                              void* d_out, int out_size, void* d_ws, size_t ws_size,
                              hipStream_t stream)
{
    const float* x    = (const float*)d_in[0];
    const float* Wg1  = (const float*)d_in[1];
    const float* bg1  = (const float*)d_in[2];
    const float* Wg2  = (const float*)d_in[3];
    const float* bg2  = (const float*)d_in[4];
    const float* W_in = (const float*)d_in[5];
    const float* b_in = (const float*)d_in[6];
    const float* W_md = (const float*)d_in[7];
    const float* b_md = (const float*)d_in[8];
    const float* W_ou = (const float*)d_in[9];
    const float* b_ou = (const float*)d_in[10];

    float* y    = (float*)d_out;                 // [B, OUT]
    float* loss = y + (size_t)BTOK * DOUT;       // 1 element

    char* ws = (char*)d_ws;
    size_t off = 0;
    float*  gates  = (float*)(ws + off);  off += (size_t)BTOK * NEXP * 4;  // 256 KiB
    float*  imp    = (float*)(ws + off);  off += 256;
    __bf16* Xbf    = (__bf16*)(ws + off); off += (size_t)BTOK * DIN * 2;   // 8 MiB
    __bf16* H1     = (__bf16*)(ws + off); off += (size_t)BTOK * DH * 2;    // 16 MiB
    __bf16* H2     = (__bf16*)(ws + off); off += (size_t)BTOK * DH * 2;    // 16 MiB
    __bf16* Wt_in  = (__bf16*)(ws + off); off += (size_t)NEXP * DIN * DH * 2;   // 8 MiB
    __bf16* Wt_md  = (__bf16*)(ws + off); off += (size_t)NEXP * DH * DH * 2;    // 16 MiB
    __bf16* Wt_ou  = (__bf16*)(ws + off); off += (size_t)NEXP * DH * DOUT * 2;  // 8 MiB

    // one-time (per call) operand preparation: bf16 + weight transpose
    cvt_bf16_kernel<<<(BTOK * DIN / 4 + 255) / 256, 256, 0, stream>>>(
        x, Xbf, BTOK * DIN / 4);
    for (int e = 0; e < NEXP; ++e) {
        transpose_cvt_kernel<<<dim3(DH / 32, DIN / 32), dim3(32, 8), 0, stream>>>(
            W_in + (size_t)e * DIN * DH, Wt_in + (size_t)e * DIN * DH, DIN, DH);
        transpose_cvt_kernel<<<dim3(DH / 32, DH / 32), dim3(32, 8), 0, stream>>>(
            W_md + (size_t)e * DH * DH, Wt_md + (size_t)e * DH * DH, DH, DH);
        transpose_cvt_kernel<<<dim3(DOUT / 32, DH / 32), dim3(32, 8), 0, stream>>>(
            W_ou + (size_t)e * DH * DOUT, Wt_ou + (size_t)e * DH * DOUT, DH, DOUT);
    }

    // gating + loss path
    gating_kernel<<<BTOK / 8, 256, 0, stream>>>(x, Wg1, bg1, Wg2, bg2, gates);
    importance_kernel<<<NEXP, 256, 0, stream>>>(gates, imp);
    loss_kernel<<<1, 1, 0, stream>>>(imp, loss);

    zero_kernel<<<(BTOK * DOUT + 255) / 256, 256, 0, stream>>>(y, BTOK * DOUT);

    // experts, sequential for deterministic fp32 accumulation into y
    for (int e = 0; e < NEXP; ++e) {
        gemm_wmma<0><<<dim3(DH / 128, BTOK / 128), 256, 0, stream>>>(
            Xbf, Wt_in + (size_t)e * DIN * DH, b_in + (size_t)e * DH,
            DH, DIN, H1, nullptr, 0);
        gemm_wmma<0><<<dim3(DH / 128, BTOK / 128), 256, 0, stream>>>(
            H1, Wt_md + (size_t)e * DH * DH, b_md + (size_t)e * DH,
            DH, DH, H2, nullptr, 0);
        gemm_wmma<1><<<dim3(DOUT / 128, BTOK / 128), 256, 0, stream>>>(
            H2, Wt_ou + (size_t)e * DH * DOUT, b_ou + (size_t)e * DOUT,
            DOUT, DH, y, gates, e);
    }
}
